// DecoderBlock_39109972198039
// MI455X (gfx1250) — matrix operand order repaired and checked by reading, not compiled, not run on hardware
//
#include <hip/hip_runtime.h>
#include <hip/hip_bf16.h>
#include <stdint.h>

// ---------------- problem constants ----------------
#define Bc 2
#define Tc 512
#define Sc 512
#define Dc 1024
#define HEc 8
#define DHc 128
#define HHc 4
#define Nc 32
#define FFc 4096
#define GHc 64
#define Mrows (Bc*Tc)   // 1024 decoder rows
#define Erows (Bc*Sc)   // 1024 encoder rows

typedef __attribute__((ext_vector_type(16))) __bf16 v16bf;
typedef __attribute__((ext_vector_type(8)))  float  v8f;

__device__ __forceinline__ __bf16 f2b(float f) {
    union { float f; unsigned u; } v; v.f = f;
    unsigned r = v.u + 0x7FFFu + ((v.u >> 16) & 1u);   // round-to-nearest-even
    union { unsigned short s; __bf16 b; } o; o.s = (unsigned short)(r >> 16);
    return o.b;
}

__device__ __forceinline__ float gelu_t(float x) {
    float x3 = x * x * x;
    return 0.5f * x * (1.f + tanhf(0.7978845608028654f * (x + 0.044715f * x3)));
}

// ---------------- generic bf16 WMMA GEMM ----------------
// C[z] = act(alpha * A[z](MxK) @ B[z](KxN) + bias) (+Res), z -> (b,h) batch
// Block tile 128x128, 256 threads = 8 wave32; per-wave 32x64 accumulator
// (2x4 WMMA tiles -> 8 v_wmma_f32_16x16x32_bf16 per K-step).
struct GemmP {
    const __bf16* A; const __bf16* B; float* C; __bf16* Cb;
    const float* bias; const float* Res;
    long offAb, offAh, offBb, offBh, offCb, offCh;
    int lda, ldb, ldc, ldr, M, N, K, nH, act, wantBf, hasBias, hasRes;
    float alpha;
};

template <int TB>
__global__ __launch_bounds__(256) void db_wmma_gemm(GemmP p) {
    // row stride 40 elems = 80B keeps every 16B chunk aligned (80 % 16 == 0)
    __shared__ __align__(16) __bf16 As[128][40];
    __shared__ __align__(16) __bf16 Bs[128][40];   // stored [n][k]

    int z = blockIdx.z; int bb = z / p.nH, hh = z % p.nH;
    const __bf16* A  = p.A + (long)bb * p.offAb + (long)hh * p.offAh;
    const __bf16* Bm = p.B + (long)bb * p.offBb + (long)hh * p.offBh;
    float* C = p.C + (long)bb * p.offCb + (long)hh * p.offCh;
    __bf16* Cb = p.wantBf ? (p.Cb + (long)bb * p.offCb + (long)hh * p.offCh) : (__bf16*)0;
    const float* Res = p.Res;

    int m0 = blockIdx.x * 128, n0 = blockIdx.y * 128;
    int tid = threadIdx.x, lane = tid & 31, wave = tid >> 5;
    int wm = (wave >> 1) * 32;      // 4 wave-rows
    int wn = (wave & 1) * 64;       // 2 wave-cols

    v8f zero = {0.f,0.f,0.f,0.f,0.f,0.f,0.f,0.f};
    v8f acc[2][4];
    #pragma unroll
    for (int i = 0; i < 2; ++i)
        #pragma unroll
        for (int j = 0; j < 4; ++j) acc[i][j] = zero;

    for (int k0 = 0; k0 < p.K; k0 += 32) {
        { // A tile 128x32: 16 bf16 per thread, two b128 per thread
            int r = tid >> 1, c = (tid & 1) * 16;
            const __bf16* src = A + (long)(m0 + r) * p.lda + (k0 + c);
            uint4 u0 = *(const uint4*)src;
            uint4 u1 = *(const uint4*)(src + 8);
            *(uint4*)&As[r][c]     = u0;
            *(uint4*)&As[r][c + 8] = u1;
            if (k0 + 32 < p.K) __builtin_prefetch((const void*)(src + 32), 0, 1);
        }
        if (TB) { // B given as N x K row-major -> direct contiguous copy
            int r = tid >> 1, c = (tid & 1) * 16;
            const __bf16* src = Bm + (long)(n0 + r) * p.ldb + (k0 + c);
            uint4 u0 = *(const uint4*)src;
            uint4 u1 = *(const uint4*)(src + 8);
            *(uint4*)&Bs[r][c]     = u0;
            *(uint4*)&Bs[r][c + 8] = u1;
            if (k0 + 32 < p.K) __builtin_prefetch((const void*)(src + 32), 0, 1);
        } else {  // B given as K x N row-major -> transpose into LDS
            int kk = tid >> 3, nc = (tid & 7) * 16;
            const __bf16* src = Bm + (long)(k0 + kk) * p.ldb + (n0 + nc);
            #pragma unroll
            for (int i = 0; i < 16; ++i) Bs[nc + i][kk] = src[i];
        }
        __syncthreads();

        // fragments per documented 16-bit A(16x32)/B(32x16) wave32 layouts
        union FA { uint4 u[2]; v16bf v; };
        FA fa[2], fb[4];
        int rl = lane & 15, hs = lane >> 4;
        #pragma unroll
        for (int i = 0; i < 2; ++i) {
            int r = wm + i * 16 + rl;
            fa[i].u[0] = *(const uint4*)&As[r][hs * 8];
            fa[i].u[1] = *(const uint4*)&As[r][16 + hs * 8];
        }
        #pragma unroll
        for (int j = 0; j < 4; ++j) {
            int c = wn + j * 16 + rl;
            fb[j].u[0] = *(const uint4*)&Bs[c][hs * 8];
            fb[j].u[1] = *(const uint4*)&Bs[c][16 + hs * 8];
        }
        #pragma unroll
        for (int i = 0; i < 2; ++i)
            #pragma unroll
            for (int j = 0; j < 4; ++j)
                acc[i][j] = __builtin_amdgcn_wmma_f32_16x16x32_bf16(
                    false, fa[i].v, false, fb[j].v, (short)0, acc[i][j], false, false);
        __syncthreads();
    }

    // epilogue, f32 C/D layout: VGPR r -> M = r + (lane>>4)*8, N = lane&15
    // all launch shapes are exact multiples of the 128x128 tile -> no guards
    int rb = (lane >> 4) * 8, cc = lane & 15;
    #pragma unroll
    for (int i = 0; i < 2; ++i)
    #pragma unroll
    for (int j = 0; j < 4; ++j) {
        int gn = n0 + wn + j * 16 + cc;
        #pragma unroll
        for (int r = 0; r < 8; ++r) {
            int gm = m0 + wm + i * 16 + rb + r;
            float v = acc[i][j][r] * p.alpha;
            if (p.hasBias) v += p.bias[gn];
            if (p.act == 1) v = gelu_t(v);
            if (p.hasRes)  v += Res[(long)gm * p.ldr + gn];
            C[(long)gm * p.ldc + gn] = v;
            if (p.wantBf) Cb[(long)gm * p.ldc + gn] = f2b(v);
        }
    }
}

// ---------------- elementwise / reduction kernels ----------------
__global__ void db_convert(const float* x, __bf16* o, long n) {
    long i = (long)blockIdx.x * 256 + threadIdx.x;
    if (i < n) o[i] = f2b(x[i]);
}

__global__ void db_layernorm(const float* x, const float* sc, const float* bi,
                             float* o, __bf16* ob) {
    int row = blockIdx.x, tid = threadIdx.x;
    const float* xr = x + (long)row * Dc;
    float s = 0.f, s2 = 0.f;
    for (int i = tid; i < Dc; i += 256) { float v = xr[i]; s += v; s2 += v * v; }
    for (int off = 16; off > 0; off >>= 1) { s += __shfl_down(s, off); s2 += __shfl_down(s2, off); }
    __shared__ float r1[8], r2[8];
    if ((tid & 31) == 0) { r1[tid >> 5] = s; r2[tid >> 5] = s2; }
    __syncthreads();
    if (tid == 0) {
        float t1 = 0.f, t2 = 0.f;
        for (int i = 0; i < 8; ++i) { t1 += r1[i]; t2 += r2[i]; }
        float m = t1 / Dc;
        r1[0] = m; r2[0] = t2 / Dc - m * m;
    }
    __syncthreads();
    float mean = r1[0], rstd = rsqrtf(r2[0] + 1e-5f);
    for (int i = tid; i < Dc; i += 256) {
        float v = (xr[i] - mean) * rstd * sc[i] + bi[i];
        o[(long)row * Dc + i] = v;
        ob[(long)row * Dc + i] = f2b(v);
    }
}

__global__ void db_rope128(const float* x, __bf16* o, int rows, int T_) {
    long idx = (long)blockIdx.x * 256 + threadIdx.x;
    if (idx >= (long)rows * Dc) return;
    int d = (int)(idx & (Dc - 1));
    long m = idx >> 10;
    int t = (int)(m % T_);
    int dd = d & (DHc - 1);
    int i = dd & 63;
    float ang = (float)t * __powf(10000.f, -(float)(2 * i) * (1.f / 128.f));
    float c = __cosf(ang), s = __sinf(ang);
    float xv = x[idx];
    float xr = (dd < 64) ? -x[idx + 64] : x[idx - 64];
    o[idx] = f2b(xv * c + xr * s);
}

__global__ void db_softmax(const float* sc, const unsigned char* mask,
                           float* pf, __bf16* pb, int Tq, int L) {
    long row = blockIdx.x; int tid = threadIdx.x;
    const float* sr = sc + row * (long)L;
    int q = (int)(row % Tq);
    const unsigned char* mr = mask + (long)q * L;
    __shared__ float r1[8];
    float mx = -3.0e38f;
    for (int i = tid; i < L; i += 256) {
        float v = mr[i] ? sr[i] : -1.0e30f;
        mx = fmaxf(mx, v);
    }
    for (int off = 16; off > 0; off >>= 1) mx = fmaxf(mx, __shfl_down(mx, off));
    if ((tid & 31) == 0) r1[tid >> 5] = mx;
    __syncthreads();
    if (tid == 0) { float t = r1[0]; for (int i = 1; i < 8; ++i) t = fmaxf(t, r1[i]); r1[0] = t; }
    __syncthreads();
    mx = r1[0];
    __syncthreads();
    float sum = 0.f;
    for (int i = tid; i < L; i += 256) {
        float v = mr[i] ? sr[i] : -1.0e30f;
        sum += __expf(v - mx);
    }
    for (int off = 16; off > 0; off >>= 1) sum += __shfl_down(sum, off);
    if ((tid & 31) == 0) r1[tid >> 5] = sum;
    __syncthreads();
    if (tid == 0) { float t = 0.f; for (int i = 0; i < 8; ++i) t += r1[i]; r1[0] = t; }
    __syncthreads();
    float inv = 1.f / r1[0];
    for (int i = tid; i < L; i += 256) {
        float v = mr[i] ? sr[i] : -1.0e30f;
        float pv = __expf(v - mx) * inv;
        pf[row * (long)L + i] = pv;
        pb[row * (long)L + i] = f2b(pv);
    }
}

// ---------------- Lorentz branch ----------------
// rope(N=32) + hyperboloid lift: (b,h,t,33) layouts
__global__ void db_lprep(const float* uq, const float* uk, const float* uv,
                         float* qh, float* kh, float* vh, int rows, int T_) {
    int gid = blockIdx.x * 256 + threadIdx.x;
    if (gid >= rows * HHc) return;
    int m = gid >> 2, h = gid & 3;
    int b = m / T_, t = m % T_;
    const float* q = uq + (long)m * (HHc * Nc) + h * Nc;
    const float* k = uk + (long)m * (HHc * Nc) + h * Nc;
    const float* v = uv + (long)m * (HHc * Nc) + h * Nc;
    float rq[32], rk[32];
    float sq = 0.f, sk = 0.f, sv = 0.f;
    #pragma unroll
    for (int d = 0; d < 32; ++d) {
        int i = d & 15;
        float ang = (float)t * __powf(10000.f, -(float)(2 * i) * (1.f / 32.f));
        float c = __cosf(ang), s = __sinf(ang);
        float qr = (d < 16) ? -q[d + 16] : q[d - 16];
        float kr = (d < 16) ? -k[d + 16] : k[d - 16];
        rq[d] = q[d] * c + qr * s;
        rk[d] = k[d] * c + kr * s;
        sq += rq[d] * rq[d]; sk += rk[d] * rk[d];
        float vv = v[d]; sv += vv * vv;
    }
    long base = (((long)b * HHc + h) * T_ + t) * 33;
    qh[base] = sqrtf(1.f + sq);
    kh[base] = sqrtf(1.f + sk);
    vh[base] = sqrtf(1.f + sv);
    #pragma unroll
    for (int d = 0; d < 32; ++d) {
        qh[base + 1 + d] = rq[d];
        kh[base + 1 + d] = rk[d];
        vh[base + 1 + d] = v[d];
    }
}

__global__ void db_lscore(const float* qh, const float* kh, float* sc,
                          int Tq, int Tk, long total) {
    long idx = (long)blockIdx.x * 256 + threadIdx.x;
    if (idx >= total) return;
    int k = (int)(idx % Tk);
    long r = idx / Tk;          // bh*Tq + q
    long bh = r / Tq;
    const float* Q = qh + r * 33;
    const float* K = kh + (bh * Tk + k) * 33;
    float inner = -Q[0] * K[0];
    #pragma unroll
    for (int d = 1; d < 33; ++d) inner += Q[d] * K[d];
    float a = fmaxf(-inner, 1.f + 1e-7f);
    sc[idx] = -acoshf(a);       // TAU = 1
}

// one Frechet (Karcher) step + expmap + logmap from origin, merged output (ld=132)
__global__ void db_frechet(const float* qh, const float* vh, const float* pr,
                           float* out, int Tq, int Tk) {
    int r = blockIdx.x * 256 + threadIdx.x;
    int total = Bc * HHc * Tq;
    if (r >= total) return;
    long bh = (long)r / Tq; int t = r % Tq;
    int h = (int)(bh & 3), b = (int)(bh >> 2);
    float x[33];
    const float* xq = qh + (long)r * 33;
    #pragma unroll
    for (int d = 0; d < 33; ++d) x[d] = xq[d];
    float tv[33];
    #pragma unroll
    for (int d = 0; d < 33; ++d) tv[d] = 0.f;
    float sA = 0.f;
    const float* p = pr + (long)r * Tk;
    for (int k = 0; k < Tk; ++k) {
        const float* v = vh + (bh * Tk + k) * 33;
        float inner = -x[0] * v[0];
        #pragma unroll
        for (int d = 1; d < 33; ++d) inner += x[d] * v[d];
        float a = fmaxf(-inner, 1.f + 1e-7f);
        float coef = acoshf(a) * rsqrtf(fmaxf(a * a - 1.f, 1e-12f));
        float w = p[k] * coef;
        #pragma unroll
        for (int d = 0; d < 33; ++d) tv[d] += w * v[d];
        sA += w * a;
    }
    float tt[33];
    #pragma unroll
    for (int d = 0; d < 33; ++d) tt[d] = tv[d] - sA * x[d];
    float nn = -tt[0] * tt[0];
    #pragma unroll
    for (int d = 1; d < 33; ++d) nn += tt[d] * tt[d];
    float n = sqrtf(fmaxf(nn, 0.f) + 1e-12f);
    float ch = coshf(n), sh = sinhf(n) / n;
    float mu[33];
    #pragma unroll
    for (int d = 0; d < 33; ++d) mu[d] = ch * x[d] + sh * tt[d];
    float a0 = fmaxf(mu[0], 1.f + 1e-7f);
    float coef = acoshf(a0) * rsqrtf(fmaxf(a0 * a0 - 1.f, 1e-12f));
    float* o = out + ((long)(b * Tq + t)) * (HHc * 33) + h * 33;
    o[0] = coef * (mu[0] - a0);
    #pragma unroll
    for (int d = 1; d < 33; ++d) o[d] = coef * mu[d];
}

// fp32 fallback GEMM for K=132 Lorentz output projection
__global__ void db_sgemm(const float* A, const float* W, const float* bias,
                         float* C, __bf16* Cb, int M_, int N_, int K_) {
    long idx = (long)blockIdx.x * 256 + threadIdx.x;
    if (idx >= (long)M_ * N_) return;
    int n = (int)(idx % N_); long m = idx / N_;
    float s = bias[n];
    const float* a = A + m * K_;
    for (int k = 0; k < K_; ++k) s += a[k] * W[(long)k * N_ + n];
    C[idx] = s;
    Cb[idx] = f2b(s);
}

// ---------------- fusion gate ----------------
__global__ void db_gate_g(const float* ee, const float* hh, const float* W,
                          const float* b, float* g) {
    int idx = blockIdx.x * 256 + threadIdx.x;
    if (idx >= Mrows * GHc) return;
    int n = idx & (GHc - 1);
    int m = idx >> 6;
    float s = b[n];
    const float* er = ee + (long)m * Dc;
    const float* hr = hh + (long)m * Dc;
    for (int k = 0; k < Dc; ++k) s += er[k] * W[(long)k * GHc + n];
    for (int k = 0; k < Dc; ++k) s += hr[k] * W[(long)(Dc + k) * GHc + n];
    g[idx] = tanhf(s);
}

__global__ void db_gate_al(const float* g, const float* w, const float* b, float* al) {
    int m = blockIdx.x * 256 + threadIdx.x;
    if (m >= Mrows) return;
    float s = b[0];
    #pragma unroll
    for (int k = 0; k < GHc; ++k) s += g[m * GHc + k] * w[k];
    al[m] = 1.f / (1.f + __expf(-s));
}

__global__ void db_fuse_add(const float* y, const float* ee, const float* hh,
                            const float* al, float* o, long n) {
    long i = (long)blockIdx.x * 256 + threadIdx.x;
    if (i >= n) return;
    long m = i >> 10;
    float a = al[m];
    o[i] = y[i] + a * hh[i] + (1.f - a) * ee[i];
}

// ---------------- host orchestration ----------------
extern "C" void kernel_launch(void* const* d_in, const int* in_sizes, int n_in,
                              void* d_out, int out_size, void* d_ws, size_t ws_size,
                              hipStream_t stream) {
    (void)in_sizes; (void)n_in; (void)out_size; (void)ws_size;
    hipStream_t st = stream;

    const float* y   = (const float*)d_in[0];
    const float* enc = (const float*)d_in[1];
    const unsigned char* smask = (const unsigned char*)d_in[2];
    const unsigned char* cmask = (const unsigned char*)d_in[3];
    const float *se_qw=(const float*)d_in[4],  *se_qb=(const float*)d_in[5];
    const float *se_kw=(const float*)d_in[6],  *se_kb=(const float*)d_in[7];
    const float *se_vw=(const float*)d_in[8],  *se_vb=(const float*)d_in[9];
    const float *se_ow=(const float*)d_in[10], *se_ob=(const float*)d_in[11];
    const float *ce_qw=(const float*)d_in[12], *ce_qb=(const float*)d_in[13];
    const float *ce_kw=(const float*)d_in[14], *ce_kb=(const float*)d_in[15];
    const float *ce_vw=(const float*)d_in[16], *ce_vb=(const float*)d_in[17];
    const float *ce_ow=(const float*)d_in[18], *ce_ob=(const float*)d_in[19];
    const float *sl_qw=(const float*)d_in[20], *sl_qb=(const float*)d_in[21];
    const float *sl_kw=(const float*)d_in[22], *sl_kb=(const float*)d_in[23];
    const float *sl_vw=(const float*)d_in[24], *sl_vb=(const float*)d_in[25];
    const float *sl_ow=(const float*)d_in[26], *sl_ob=(const float*)d_in[27];
    const float *cl_qw=(const float*)d_in[28], *cl_qb=(const float*)d_in[29];
    const float *cl_kw=(const float*)d_in[30], *cl_kb=(const float*)d_in[31];
    const float *cl_vw=(const float*)d_in[32], *cl_vb=(const float*)d_in[33];
    const float *cl_ow=(const float*)d_in[34], *cl_ob=(const float*)d_in[35];
    const float *fs_Uew=(const float*)d_in[36], *fs_Ueb=(const float*)d_in[37];
    const float *fs_Uhw=(const float*)d_in[38], *fs_Uhb=(const float*)d_in[39];
    const float *fs_ghw=(const float*)d_in[40], *fs_ghb=(const float*)d_in[41];
    const float *fs_gow=(const float*)d_in[42], *fs_gob=(const float*)d_in[43];
    const float *fc_Uew=(const float*)d_in[44], *fc_Ueb=(const float*)d_in[45];
    const float *fc_Uhw=(const float*)d_in[46], *fc_Uhb=(const float*)d_in[47];
    const float *fc_ghw=(const float*)d_in[48], *fc_ghb=(const float*)d_in[49];
    const float *fc_gow=(const float*)d_in[50], *fc_gob=(const float*)d_in[51];
    const float *lns_s=(const float*)d_in[52], *lns_b=(const float*)d_in[53];
    const float *lnc_s=(const float*)d_in[54], *lnc_b=(const float*)d_in[55];
    const float *lnf_s=(const float*)d_in[56], *lnf_b=(const float*)d_in[57];
    const float *fiw=(const float*)d_in[58], *fib=(const float*)d_in[59];
    const float *fow=(const float*)d_in[60], *fob=(const float*)d_in[61];

    // ---- workspace bump allocator ----
    char* base = (char*)d_ws;
    size_t off = 0;
    auto alloc = [&](size_t bytes) -> void* {
        void* p = base + off;
        off += (bytes + 255) & ~(size_t)255;
        return p;
    };
    const size_t MD  = (size_t)Mrows * Dc;             // 1M
    const size_t SCN = (size_t)Bc * HEc * Tc * Sc;     // 4M euclid score elems

    // bf16 weights
    __bf16* wb_se_q = (__bf16*)alloc(MD * 2); __bf16* wb_se_k = (__bf16*)alloc(MD * 2);
    __bf16* wb_se_v = (__bf16*)alloc(MD * 2); __bf16* wb_se_o = (__bf16*)alloc(MD * 2);
    __bf16* wb_ce_q = (__bf16*)alloc(MD * 2); __bf16* wb_ce_k = (__bf16*)alloc(MD * 2);
    __bf16* wb_ce_v = (__bf16*)alloc(MD * 2); __bf16* wb_ce_o = (__bf16*)alloc(MD * 2);
    __bf16* wb_sl_q = (__bf16*)alloc((size_t)Dc * 128 * 2);
    __bf16* wb_sl_k = (__bf16*)alloc((size_t)Dc * 128 * 2);
    __bf16* wb_sl_v = (__bf16*)alloc((size_t)Dc * 128 * 2);
    __bf16* wb_cl_q = (__bf16*)alloc((size_t)Dc * 128 * 2);
    __bf16* wb_cl_k = (__bf16*)alloc((size_t)Dc * 128 * 2);
    __bf16* wb_cl_v = (__bf16*)alloc((size_t)Dc * 128 * 2);
    __bf16* wb_fs_Ue = (__bf16*)alloc(MD * 2); __bf16* wb_fs_Uh = (__bf16*)alloc(MD * 2);
    __bf16* wb_fc_Ue = (__bf16*)alloc(MD * 2); __bf16* wb_fc_Uh = (__bf16*)alloc(MD * 2);
    __bf16* wb_fi = (__bf16*)alloc((size_t)Dc * FFc * 2);
    __bf16* wb_fo = (__bf16*)alloc((size_t)FFc * Dc * 2);
    __bf16* enc_b = (__bf16*)alloc(MD * 2);

    // activations
    float*  hs_f = (float*)alloc(MD * 4);   __bf16* hs_b = (__bf16*)alloc(MD * 2);
    float*  qe_f = (float*)alloc(MD * 4);   float* ke_f = (float*)alloc(MD * 4);
    float*  ve_f = (float*)alloc(MD * 4);   __bf16* ve_b = (__bf16*)alloc(MD * 2);
    __bf16* qe_b = (__bf16*)alloc(MD * 2);  __bf16* ke_b = (__bf16*)alloc(MD * 2);
    float*  sc_f = (float*)alloc(SCN * 4);
    float*  pr_f = (float*)alloc(SCN * 4);  __bf16* pr_b = (__bf16*)alloc(SCN * 2);
    float*  at_f = (float*)alloc(MD * 4);   __bf16* at_b = (__bf16*)alloc(MD * 2);
    float*  e_f  = (float*)alloc(MD * 4);   __bf16* e_b  = (__bf16*)alloc(MD * 2);
    float*  ulq  = (float*)alloc((size_t)Mrows * 128 * 4);
    float*  ulk  = (float*)alloc((size_t)Mrows * 128 * 4);
    float*  ulv  = (float*)alloc((size_t)Mrows * 128 * 4);
    float*  qh_  = (float*)alloc((size_t)Bc * HHc * Tc * 33 * 4);
    float*  kh_  = (float*)alloc((size_t)Bc * HHc * Sc * 33 * 4);
    float*  vh_  = (float*)alloc((size_t)Bc * HHc * Sc * 33 * 4);
    float*  ylog = (float*)alloc((size_t)Mrows * 132 * 4);
    float*  h_f  = (float*)alloc(MD * 4);   __bf16* h_b = (__bf16*)alloc(MD * 2);
    float*  ee_f = (float*)alloc(MD * 4);   float* hh_f = (float*)alloc(MD * 4);
    float*  g_   = (float*)alloc((size_t)Mrows * GHc * 4);
    float*  al_  = (float*)alloc((size_t)Mrows * 4);
    float*  y1   = (float*)alloc(MD * 4);   float* y2 = (float*)alloc(MD * 4);
    float*  f1_f = (float*)alloc((size_t)Mrows * FFc * 4);
    __bf16* f1_b = (__bf16*)alloc((size_t)Mrows * FFc * 2);
    // Lorentz score/prob buffers alias the (larger, already-consumed) euclid ones
    float*  lsc  = sc_f;  float* lpr_f = pr_f;  __bf16* lpr_b = pr_b;

    auto conv = [&](const float* src, __bf16* dst, long n) {
        db_convert<<<dim3((unsigned)((n + 255) / 256)), dim3(256), 0, st>>>(src, dst, n);
    };
    auto gemm = [&](const __bf16* A, long oAb, long oAh, int lda,
                    const __bf16* Bm, long oBb, long oBh, int ldb, int tB,
                    float* C, __bf16* Cb, long oCb, long oCh, int ldc,
                    const float* bias, const float* Res, int ldr,
                    int M, int N, int K, int nB, int nH, float alpha, int act) {
        GemmP p;
        p.A = A; p.B = Bm; p.C = C; p.Cb = Cb; p.bias = bias; p.Res = Res;
        p.offAb = oAb; p.offAh = oAh; p.offBb = oBb; p.offBh = oBh;
        p.offCb = oCb; p.offCh = oCh;
        p.lda = lda; p.ldb = ldb; p.ldc = ldc; p.ldr = ldr;
        p.M = M; p.N = N; p.K = K; p.nH = nH; p.act = act;
        p.wantBf = (Cb != nullptr); p.hasBias = (bias != nullptr); p.hasRes = (Res != nullptr);
        p.alpha = alpha;
        dim3 grid((M + 127) / 128, (N + 127) / 128, nB * nH);
        if (tB) db_wmma_gemm<1><<<grid, dim3(256), 0, st>>>(p);
        else    db_wmma_gemm<0><<<grid, dim3(256), 0, st>>>(p);
    };

    // ---- weight conversion (once per call) ----
    conv(se_qw, wb_se_q, MD); conv(se_kw, wb_se_k, MD); conv(se_vw, wb_se_v, MD); conv(se_ow, wb_se_o, MD);
    conv(ce_qw, wb_ce_q, MD); conv(ce_kw, wb_ce_k, MD); conv(ce_vw, wb_ce_v, MD); conv(ce_ow, wb_ce_o, MD);
    conv(sl_qw, wb_sl_q, (long)Dc * 128); conv(sl_kw, wb_sl_k, (long)Dc * 128); conv(sl_vw, wb_sl_v, (long)Dc * 128);
    conv(cl_qw, wb_cl_q, (long)Dc * 128); conv(cl_kw, wb_cl_k, (long)Dc * 128); conv(cl_vw, wb_cl_v, (long)Dc * 128);
    conv(fs_Uew, wb_fs_Ue, MD); conv(fs_Uhw, wb_fs_Uh, MD);
    conv(fc_Uew, wb_fc_Ue, MD); conv(fc_Uhw, wb_fc_Uh, MD);
    conv(fiw, wb_fi, (long)Dc * FFc); conv(fow, wb_fo, (long)FFc * Dc);
    conv(enc, enc_b, MD);

    const float rscale = 0.08838834764831845f;  // 1/sqrt(128)

    auto euclid_attn = [&](const __bf16* xq_b, const __bf16* xkv_b,
                           const __bf16* wq, const float* bq,
                           const __bf16* wk, const float* bk,
                           const __bf16* wv, const float* bv,
                           const __bf16* wo, const float* bo,
                           const unsigned char* mask) {
        gemm(xq_b,0,0,Dc,  wq,0,0,Dc,0,  qe_f,nullptr,0,0,Dc, bq,nullptr,0, Mrows,Dc,Dc,1,1,1.f,0);
        gemm(xkv_b,0,0,Dc, wk,0,0,Dc,0,  ke_f,nullptr,0,0,Dc, bk,nullptr,0, Erows,Dc,Dc,1,1,1.f,0);
        gemm(xkv_b,0,0,Dc, wv,0,0,Dc,0,  ve_f,ve_b,0,0,Dc,    bv,nullptr,0, Erows,Dc,Dc,1,1,1.f,0);
        db_rope128<<<dim3(Mrows * Dc / 256), 256, 0, st>>>(qe_f, qe_b, Mrows, Tc);
        db_rope128<<<dim3(Erows * Dc / 256), 256, 0, st>>>(ke_f, ke_b, Erows, Sc);
        // scores: per (b,h): (T x dh) @ (dh x S) via transB
        gemm(qe_b,(long)Tc*Dc,DHc,Dc,  ke_b,(long)Sc*Dc,DHc,Dc,1,
             sc_f,nullptr,(long)HEc*Tc*Sc,(long)Tc*Sc,Sc, nullptr,nullptr,0,
             Tc,Sc,DHc, Bc,HEc, rscale, 0);
        db_softmax<<<dim3(Bc * HEc * Tc), 256, 0, st>>>(sc_f, mask, pr_f, pr_b, Tc, Sc);
        // AV: per (b,h): (T x S) @ (S x dh), merged output
        gemm(pr_b,(long)HEc*Tc*Sc,(long)Tc*Sc,Sc,  ve_b,(long)Sc*Dc,DHc,Dc,0,
             at_f,at_b,(long)Tc*Dc,DHc,Dc, nullptr,nullptr,0,
             Tc,DHc,Sc, Bc,HEc, 1.f, 0);
        gemm(at_b,0,0,Dc, wo,0,0,Dc,0, e_f,e_b,0,0,Dc, bo,nullptr,0, Mrows,Dc,Dc,1,1,1.f,0);
    };

    auto lorentz_attn = [&](const __bf16* xq_b, const __bf16* xkv_b,
                            const __bf16* wq, const float* bq,
                            const __bf16* wk, const float* bk,
                            const __bf16* wv, const float* bv,
                            const float* wo, const float* bo,
                            const unsigned char* mask) {
        gemm(xq_b,0,0,Dc,  wq,0,0,128,0, ulq,nullptr,0,0,128, bq,nullptr,0, Mrows,128,Dc,1,1,1.f,0);
        gemm(xkv_b,0,0,Dc, wk,0,0,128,0, ulk,nullptr,0,0,128, bk,nullptr,0, Erows,128,Dc,1,1,1.f,0);
        gemm(xkv_b,0,0,Dc, wv,0,0,128,0, ulv,nullptr,0,0,128, bv,nullptr,0, Erows,128,Dc,1,1,1.f,0);
        db_lprep<<<dim3((Mrows * HHc + 255) / 256), 256, 0, st>>>(ulq, ulk, ulv, qh_, kh_, vh_, Mrows, Tc);
        long totS = (long)Bc * HHc * Tc * Sc;
        db_lscore<<<dim3((unsigned)((totS + 255) / 256)), 256, 0, st>>>(qh_, kh_, lsc, Tc, Sc, totS);
        db_softmax<<<dim3(Bc * HHc * Tc), 256, 0, st>>>(lsc, mask, lpr_f, lpr_b, Tc, Sc);
        db_frechet<<<dim3((Bc * HHc * Tc + 255) / 256), 256, 0, st>>>(qh_, vh_, lpr_f, ylog, Tc, Sc);
        db_sgemm<<<dim3((unsigned)((MD + 255) / 256)), 256, 0, st>>>(ylog, wo, bo, h_f, h_b, Mrows, Dc, 132);
    };

    auto fuse = [&](const __bf16* wUe, const float* bUe,
                    const __bf16* wUh, const float* bUh,
                    const float* wg, const float* bg,
                    const float* wgo, const float* bgo,
                    const float* yin, float* yout) {
        gemm(e_b,0,0,Dc, wUe,0,0,Dc,0, ee_f,nullptr,0,0,Dc, bUe,nullptr,0, Mrows,Dc,Dc,1,1,1.f,0);
        gemm(h_b,0,0,Dc, wUh,0,0,Dc,0, hh_f,nullptr,0,0,Dc, bUh,nullptr,0, Mrows,Dc,Dc,1,1,1.f,0);
        db_gate_g<<<dim3((Mrows * GHc + 255) / 256), 256, 0, st>>>(ee_f, hh_f, wg, bg, g_);
        db_gate_al<<<dim3((Mrows + 255) / 256), 256, 0, st>>>(g_, wgo, bgo, al_);
        db_fuse_add<<<dim3((unsigned)(MD / 256)), 256, 0, st>>>(yin, ee_f, hh_f, al_, yout, (long)MD);
    };

    // ---- self block ----
    db_layernorm<<<dim3(Mrows), 256, 0, st>>>(y, lns_s, lns_b, hs_f, hs_b);
    euclid_attn(hs_b, hs_b, wb_se_q, se_qb, wb_se_k, se_kb, wb_se_v, se_vb, wb_se_o, se_ob, smask);
    lorentz_attn(hs_b, hs_b, wb_sl_q, sl_qb, wb_sl_k, sl_kb, wb_sl_v, sl_vb, sl_ow, sl_ob, smask);
    fuse(wb_fs_Ue, fs_Ueb, wb_fs_Uh, fs_Uhb, fs_ghw, fs_ghb, fs_gow, fs_gob, y, y1);

    // ---- cross block ----
    db_layernorm<<<dim3(Mrows), 256, 0, st>>>(y1, lnc_s, lnc_b, hs_f, hs_b);
    euclid_attn(hs_b, enc_b, wb_ce_q, ce_qb, wb_ce_k, ce_kb, wb_ce_v, ce_vb, wb_ce_o, ce_ob, cmask);
    lorentz_attn(hs_b, enc_b, wb_cl_q, cl_qb, wb_cl_k, cl_kb, wb_cl_v, cl_vb, cl_ow, cl_ob, cmask);
    fuse(wb_fc_Ue, fc_Ueb, wb_fc_Uh, fc_Uhb, fc_ghw, fc_ghb, fc_gow, fc_gob, y1, y2);

    // ---- FFN ----
    db_layernorm<<<dim3(Mrows), 256, 0, st>>>(y2, lnf_s, lnf_b, hs_f, hs_b);
    gemm(hs_b,0,0,Dc, wb_fi,0,0,FFc,0, f1_f,f1_b,0,0,FFc, fib,nullptr,0, Mrows,FFc,Dc,1,1,1.f,1 /*gelu*/);
    gemm(f1_b,0,0,FFc, wb_fo,0,0,Dc,0, (float*)d_out,nullptr,0,0,Dc, fob, y2, Dc, Mrows,Dc,FFc,1,1,1.f,0);
}
